// ExpertParallelMoE_45681272160504
// MI455X (gfx1250) — compile-verified, matrix operand order checked by reading, not run on hardware
//
#include <hip/hip_runtime.h>

// ---------------------------------------------------------------------------
// MoE forward (top-1, E=8, D=1024, H=4096, C=2048, N=8192) for gfx1250.
// bf16 WMMA (v_wmma_f32_16x16x32_bf16), f32 accumulate, double-buffered LDS,
// async global->LDS DMA (ASYNCcnt) for the bf16 activation tiles in GEMM2.
// Native __bf16 converts (hardware cvt) for all f32->bf16 packing.
// ---------------------------------------------------------------------------

#define D_MODEL  1024
#define HID      4096
#define NEXP     8
#define CAP      2048
#define NTOK     8192

#define BM 128
#define BN 128
#define BK 32

typedef __attribute__((ext_vector_type(16))) __bf16 v16bf;
typedef __attribute__((ext_vector_type(8)))  float  v8f;
typedef __attribute__((ext_vector_type(4)))  int    v4i;

#if __has_builtin(__builtin_amdgcn_global_load_async_to_lds_b128)
#define HAVE_ASYNC_LDS 1
#else
#define HAVE_ASYNC_LDS 0
#endif

#define AS1 __attribute__((address_space(1)))
#define AS3 __attribute__((address_space(3)))

#if HAVE_ASYNC_LDS
__device__ __forceinline__ void async_copy16(const void* g, void* l) {
    // 16 bytes per lane, global -> LDS, tracked with ASYNCcnt
    __builtin_amdgcn_global_load_async_to_lds_b128((AS1 v4i*)g, (AS3 v4i*)l, 0, 0);
}
__device__ __forceinline__ void wait_async0() {
#if __has_builtin(__builtin_amdgcn_s_wait_asynccnt)
    __builtin_amdgcn_s_wait_asynccnt(0);
#else
    asm volatile("s_wait_asynccnt 0" ::: "memory");
#endif
}
#endif

// f32 -> bf16 via native fptrunc (RNE); lowers to hardware cvt on gfx1250
__device__ __forceinline__ unsigned short f2bf(float f) {
    union { __bf16 b; unsigned short s; } r;
    r.b = (__bf16)f;
    return r.s;
}
__device__ __forceinline__ unsigned pack_bf2(float lo, float hi) {
    union { __bf16 b[2]; unsigned u; } r;
    r.b[0] = (__bf16)lo;
    r.b[1] = (__bf16)hi;
    return r.u;
}

// Load one 16x32 bf16 WMMA fragment from an LDS tile stored [row][BK].
// ISA layout (16-bit A/B, wave32): lane L (0-15) = row L, K {0..7,16..23};
// lane L+16 = row L, K {8..15,24..31}; each VGPR = 2 packed bf16.
__device__ __forceinline__ v16bf load_frag(const unsigned short* tile_row0, int lane) {
    union { unsigned u[8]; v16bf v; } r;
    const int row = lane & 15;
    const int h   = lane >> 4;
    const unsigned short* p = tile_row0 + row * BK;
#pragma unroll
    for (int i = 0; i < 8; ++i) {
        const int kk = ((i & 4) ? 16 : 0) + h * 8 + (i & 3) * 2; // even -> 4B aligned
        r.u[i] = *(const unsigned*)(p + kk);
    }
    return r.v;
}

__device__ __forceinline__ v8f wmma_bf16(v16bf a, v16bf b, v8f c) {
    return __builtin_amdgcn_wmma_f32_16x16x32_bf16(
        false, a, false, b, (short)0, c, false, false);
}

// ---------------------------------------------------------------------------
// 1) init: zero output, tok_of = -1
// ---------------------------------------------------------------------------
__global__ __launch_bounds__(256) void moe_init_kernel(float* __restrict__ out,
                                                       int* __restrict__ tok_of) {
    const size_t i = (size_t)blockIdx.x * 256 + threadIdx.x;
    if (i < (size_t)NTOK * D_MODEL) out[i] = 0.0f;
    if (i < (size_t)NEXP * CAP)     tok_of[i] = -1;
}

// ---------------------------------------------------------------------------
// 2) gating: one wave per token, logits = x@Wg + bg, argmax
// ---------------------------------------------------------------------------
__global__ __launch_bounds__(256) void moe_gate_kernel(const float* __restrict__ x,
                                                       const float* __restrict__ Wg,
                                                       const float* __restrict__ bg,
                                                       int* __restrict__ route) {
    const int gtid = blockIdx.x * blockDim.x + threadIdx.x;
    const int tokn = gtid >> 5;
    const int lane = threadIdx.x & 31;
    if (tokn >= NTOK) return;

    const float* xr = x + (size_t)tokn * D_MODEL;
    float acc[NEXP];
#pragma unroll
    for (int e = 0; e < NEXP; ++e) acc[e] = 0.0f;

    for (int i = lane; i < D_MODEL; i += 32) {
        const float xv = xr[i];
        const float* wr = Wg + i * NEXP;
#pragma unroll
        for (int e = 0; e < NEXP; ++e) acc[e] += xv * wr[e];
    }
#pragma unroll
    for (int e = 0; e < NEXP; ++e) {
#pragma unroll
        for (int off = 16; off > 0; off >>= 1)
            acc[e] += __shfl_xor(acc[e], off, 32);
    }
    if (lane == 0) {
        int best = 0;
        float bv = acc[0] + bg[0];
#pragma unroll
        for (int e = 1; e < NEXP; ++e) {
            const float v = acc[e] + bg[e];
            if (v > bv) { bv = v; best = e; }   // strict > keeps lowest index
        }
        route[tokn] = best;
    }
}

// ---------------------------------------------------------------------------
// 3) routing scan: single wave32, ballot-based stable prefix per expert
// ---------------------------------------------------------------------------
__global__ void moe_route_scan_kernel(const int* __restrict__ route,
                                      int* __restrict__ tok_of) {
    __shared__ int counts[NEXP];
    const int lane = threadIdx.x;
    if (lane < NEXP) counts[lane] = 0;
    __syncthreads();
    const unsigned lt = (1u << lane) - 1u;

    for (int base = 0; base < NTOK; base += 32) {
        const int t = base + lane;
        const int e = route[t];
        int cnt[NEXP];
        int pos = 0;
#pragma unroll
        for (int ee = 0; ee < NEXP; ++ee) {
            const unsigned m = (unsigned)__ballot(e == ee);
            cnt[ee] = __popc(m);
            if (e == ee) pos = counts[ee] + __popc(m & lt);
        }
        __syncthreads();
        if (lane < NEXP) counts[lane] += cnt[lane];
        __syncthreads();
        if (pos < CAP) tok_of[e * CAP + pos] = t;
    }
}

// ---------------------------------------------------------------------------
// 4) GEMM1: h[e] = relu( gather(x) @ W1[e] + b1[e] ), bf16 out.
//    Double-buffered LDS, register-staged loads (A needs gather + cvt,
//    B needs f32->bf16 cvt), one barrier per k-step.
// ---------------------------------------------------------------------------
__global__ __launch_bounds__(256) void moe_gemm1_kernel(const float* __restrict__ x,
                                                        const float* __restrict__ W1,
                                                        const float* __restrict__ b1,
                                                        const int* __restrict__ tok_of,
                                                        unsigned short* __restrict__ h) {
    const int e   = blockIdx.z;
    const int m0  = blockIdx.y * BM;
    const int n0  = blockIdx.x * BN;
    const int tid = threadIdx.x;
    const int lane = tid & 31, wid = tid >> 5;
    const int wm = (wid & 3) * 32;   // 4 wave-rows x 32
    const int wn = (wid >> 2) * 64;  // 2 wave-cols x 64

    __shared__ unsigned short As[2][BM * BK]; // [m][k] bf16
    __shared__ unsigned short Bs[2][BN * BK]; // [n][k] bf16 (transposed)

    const float* W   = W1 + (size_t)e * D_MODEL * HID;
    const int*   tok = tok_of + e * CAP;

    // A loader: row ar (0..127), 16 f32 at col ac
    const int ar  = tid >> 1;
    const int ac  = (tid & 1) * 16;
    const int t_a = tok[m0 + ar];
    const float* arow = (t_a >= 0) ? (x + (size_t)t_a * D_MODEL + ac) : nullptr;
    // B loader: k-pair kp (k = 2*kp), 8 cols at ns*8
    const int kp = tid & 15;
    const int ns = tid >> 4;
    const float* bbase = W + (size_t)(2 * kp) * HID + n0 + ns * 8;

    unsigned apk[8];
    float    br0[8], br1[8];

    auto loadA = [&](int k0) {
        if (arow) {
            const float* s = arow + k0;
#pragma unroll
            for (int i = 0; i < 8; ++i) apk[i] = pack_bf2(s[2 * i], s[2 * i + 1]);
        } else {
#pragma unroll
            for (int i = 0; i < 8; ++i) apk[i] = 0u;
        }
    };
    auto loadB = [&](int k0) {
        const float* s = bbase + (size_t)k0 * HID;
        __builtin_prefetch(s + (size_t)2 * BK * HID, 0, 0);
#pragma unroll
        for (int i = 0; i < 8; ++i) { br0[i] = s[i]; br1[i] = s[HID + i]; }
    };
    auto storeA = [&](int buf) {
        unsigned* d = (unsigned*)&As[buf][ar * BK + ac];
#pragma unroll
        for (int i = 0; i < 8; ++i) d[i] = apk[i];
    };
    auto storeB = [&](int buf) {
#pragma unroll
        for (int i = 0; i < 8; ++i)
            *(unsigned*)&Bs[buf][(ns * 8 + i) * BK + 2 * kp] = pack_bf2(br0[i], br1[i]);
    };

    v8f acc[2][4];
#pragma unroll
    for (int i = 0; i < 2; ++i)
#pragma unroll
        for (int j = 0; j < 4; ++j)
#pragma unroll
            for (int v = 0; v < 8; ++v) acc[i][j][v] = 0.0f;

    loadA(0); loadB(0);
    storeA(0); storeB(0);
    __syncthreads();

    int cur = 0;
    for (int k0 = 0; k0 < D_MODEL; k0 += BK) {
        const bool more = (k0 + BK) < D_MODEL;
        if (more) { loadA(k0 + BK); loadB(k0 + BK); }

        v16bf afrag[2], bfrag[4];
#pragma unroll
        for (int i = 0; i < 2; ++i) afrag[i] = load_frag(&As[cur][(wm + 16 * i) * BK], lane);
#pragma unroll
        for (int j = 0; j < 4; ++j) bfrag[j] = load_frag(&Bs[cur][(wn + 16 * j) * BK], lane);
#pragma unroll
        for (int i = 0; i < 2; ++i)
#pragma unroll
            for (int j = 0; j < 4; ++j)
                acc[i][j] = wmma_bf16(afrag[i], bfrag[j], acc[i][j]);

        if (more) { storeA(cur ^ 1); storeB(cur ^ 1); }
        __syncthreads();
        cur ^= 1;
    }

    // epilogue: + b1, relu, -> bf16
    const int ln = lane & 15, lh = lane >> 4;
#pragma unroll
    for (int i = 0; i < 2; ++i) {
#pragma unroll
        for (int j = 0; j < 4; ++j) {
            const int ng   = n0 + wn + j * 16 + ln;
            const float bi = b1[e * HID + ng];
#pragma unroll
            for (int v = 0; v < 8; ++v) {
                const int m = m0 + wm + i * 16 + v + 8 * lh;
                float val = acc[i][j][v] + bi;
                val = val > 0.0f ? val : 0.0f;
                h[((size_t)e * CAP + m) * HID + ng] = f2bf(val);
            }
        }
    }
}

// ---------------------------------------------------------------------------
// 5) GEMM2: y = h[e] @ W2[e] + b2[e], scatter rows to out[token].
//    A tile is raw bf16 -> async global->LDS DMA into the next buffer while
//    WMMAs consume the current one (s_wait_asynccnt before the barrier).
// ---------------------------------------------------------------------------
__global__ __launch_bounds__(256) void moe_gemm2_kernel(const unsigned short* __restrict__ h,
                                                        const float* __restrict__ W2,
                                                        const float* __restrict__ b2,
                                                        const int* __restrict__ tok_of,
                                                        float* __restrict__ out) {
    const int e   = blockIdx.z;
    const int m0  = blockIdx.y * BM;
    const int n0  = blockIdx.x * BN;
    const int tid = threadIdx.x;
    const int lane = tid & 31, wid = tid >> 5;
    const int wm = (wid & 3) * 32;
    const int wn = (wid >> 2) * 64;

    __shared__ unsigned short As[2][BM * BK];
    __shared__ unsigned short Bs[2][BN * BK];

    const float* W   = W2 + (size_t)e * HID * D_MODEL;
    const int*   tok = tok_of + e * CAP;

    // A loader: row ar, 16 bf16 (32B) at col ac -> two b128 transfers
    const int ar = tid >> 1;
    const int ac = (tid & 1) * 16;
    const unsigned short* asrc = h + ((size_t)e * CAP + m0 + ar) * HID + ac;
    // B loader: k-pair kp, 8 cols at ns*8
    const int kp = tid & 15;
    const int ns = tid >> 4;
    const float* bbase = W + (size_t)(2 * kp) * D_MODEL + n0 + ns * 8;

    float br0[8], br1[8];
#if !HAVE_ASYNC_LDS
    unsigned apk[8];
#endif

    auto loadB = [&](int k0) {
        const float* s = bbase + (size_t)k0 * D_MODEL;
        __builtin_prefetch(s + (size_t)2 * BK * D_MODEL, 0, 0);
#pragma unroll
        for (int i = 0; i < 8; ++i) { br0[i] = s[i]; br1[i] = s[D_MODEL + i]; }
    };
    auto storeB = [&](int buf) {
#pragma unroll
        for (int i = 0; i < 8; ++i)
            *(unsigned*)&Bs[buf][(ns * 8 + i) * BK + 2 * kp] = pack_bf2(br0[i], br1[i]);
    };
    auto issueA = [&](int k0, int buf) {
#if HAVE_ASYNC_LDS
        async_copy16(asrc + k0,     &As[buf][ar * BK + ac]);
        async_copy16(asrc + k0 + 8, &As[buf][ar * BK + ac + 8]);
#else
        const unsigned* s = (const unsigned*)(asrc + k0);
#pragma unroll
        for (int i = 0; i < 8; ++i) apk[i] = s[i];
        unsigned* d = (unsigned*)&As[buf][ar * BK + ac];
#pragma unroll
        for (int i = 0; i < 8; ++i) d[i] = apk[i];
#endif
    };

    v8f acc[2][4];
#pragma unroll
    for (int i = 0; i < 2; ++i)
#pragma unroll
        for (int j = 0; j < 4; ++j)
#pragma unroll
            for (int v = 0; v < 8; ++v) acc[i][j][v] = 0.0f;

    // prologue: tile 0
    issueA(0, 0);
    loadB(0); storeB(0);
#if HAVE_ASYNC_LDS
    wait_async0();
#endif
    __syncthreads();

    int cur = 0;
    for (int k0 = 0; k0 < HID; k0 += BK) {
        const bool more = (k0 + BK) < HID;
        if (more) {
            issueA(k0 + BK, cur ^ 1);   // DMA into next buffer, overlaps WMMA
            loadB(k0 + BK);
        }

        v16bf afrag[2], bfrag[4];
#pragma unroll
        for (int i = 0; i < 2; ++i) afrag[i] = load_frag(&As[cur][(wm + 16 * i) * BK], lane);
#pragma unroll
        for (int j = 0; j < 4; ++j) bfrag[j] = load_frag(&Bs[cur][(wn + 16 * j) * BK], lane);
#pragma unroll
        for (int i = 0; i < 2; ++i)
#pragma unroll
            for (int j = 0; j < 4; ++j)
                acc[i][j] = wmma_bf16(afrag[i], bfrag[j], acc[i][j]);

        if (more) {
            storeB(cur ^ 1);
#if HAVE_ASYNC_LDS
            wait_async0();              // next-buffer DMA landed before barrier
#endif
        }
        __syncthreads();
        cur ^= 1;
    }

    // epilogue: + b2, scatter rows to out[token]
    const int ln = lane & 15, lh = lane >> 4;
#pragma unroll
    for (int i = 0; i < 2; ++i) {
#pragma unroll
        for (int v = 0; v < 8; ++v) {
            const int m = m0 + wm + i * 16 + v + 8 * lh;
            const int t = tok[m];
            if (t < 0) continue;
#pragma unroll
            for (int j = 0; j < 4; ++j) {
                const int ng = n0 + wn + j * 16 + ln;
                out[(size_t)t * D_MODEL + ng] = acc[i][j][v] + b2[e * D_MODEL + ng];
            }
        }
    }
}

// ---------------------------------------------------------------------------
// launcher
// ---------------------------------------------------------------------------
extern "C" void kernel_launch(void* const* d_in, const int* in_sizes, int n_in,
                              void* d_out, int out_size, void* d_ws, size_t ws_size,
                              hipStream_t stream) {
    (void)in_sizes; (void)n_in; (void)out_size; (void)ws_size;

    const float* x  = (const float*)d_in[0];
    const float* Wg = (const float*)d_in[1];
    const float* bg = (const float*)d_in[2];
    const float* W1 = (const float*)d_in[3];
    const float* b1 = (const float*)d_in[4];
    const float* W2 = (const float*)d_in[5];
    const float* b2 = (const float*)d_in[6];
    float* out = (float*)d_out;

    int* route  = (int*)d_ws;                                      // NTOK ints
    int* tok_of = route + NTOK;                                    // NEXP*CAP ints
    unsigned short* hbuf = (unsigned short*)(tok_of + NEXP * CAP); // bf16 h

    {
        const size_t n = (size_t)NTOK * D_MODEL;
        moe_init_kernel<<<dim3((unsigned)((n + 255) / 256)), dim3(256), 0, stream>>>(out, tok_of);
    }
    moe_gate_kernel<<<dim3(NTOK * 32 / 256), dim3(256), 0, stream>>>(x, Wg, bg, route);
    moe_route_scan_kernel<<<dim3(1), dim3(32), 0, stream>>>(route, tok_of);
    moe_gemm1_kernel<<<dim3(HID / BN, CAP / BM, NEXP), dim3(256), 0, stream>>>(
        x, W1, b1, tok_of, hbuf);
    moe_gemm2_kernel<<<dim3(D_MODEL / BN, CAP / BM, NEXP), dim3(256), 0, stream>>>(
        hbuf, W2, b2, tok_of, out);
}
